// Network_51488067945082
// MI455X (gfx1250) — matrix operand order repaired and checked by reading, not compiled, not run on hardware
//
#include <hip/hip_runtime.h>

// ---------------------------------------------------------------------------
// MI455X (gfx1250, wave32) 2-layer windowed attention.
// All matmuls: v_wmma_f32_16x16x32_f16 with hi/lo fp16 split (3 WMMAs per
// fp32 MAC chunk, ~2^-22 relative error). The fp32->fp16 hi/lo decomposition
// is hoisted OUT of all matmul loops: external inputs are pre-converted once,
// intermediate tensors are emitted as hi/lo fp16 by the producing kernel's
// epilogue. Matmul inner loops are pure b128 loads + WMMA.
// Attention P*V runs j-outer with 16 query rows held in v2f registers so the
// fp32 accumulation lowers to v_pk_fma_f32 with 8x fewer DS operations.
// ---------------------------------------------------------------------------

typedef __attribute__((ext_vector_type(16))) _Float16 v16h;
typedef __attribute__((ext_vector_type(8)))  _Float16 v8h;
typedef __attribute__((ext_vector_type(4)))  _Float16 v4h;
typedef __attribute__((ext_vector_type(8)))  float    v8f;
typedef __attribute__((ext_vector_type(2)))  float    v2f;

#define DIM   256
#define SEQ   4096
#define BATCH 4
#define WIN   32
#define KROWS 128      // key/value rows staged per 64-query block
#define KSTH  264      // fp16 LDS row stride (halves): 528 B = 4 dwords mod 64 banks
#define VST   260      // fp32 LDS row stride (floats): 4 dwords mod 64 banks

#define CAT16(a, b) __builtin_shufflevector(a, b, 0,1,2,3,4,5,6,7,8,9,10,11,12,13,14,15)

__device__ __forceinline__ v8f wmma_hilo(v16h Ah, v16h Al, v16h Bh, v16h Bl, v8f c) {
  c = __builtin_amdgcn_wmma_f32_16x16x32_f16(false, Ah, false, Bh, (short)0, c, false, false);
  c = __builtin_amdgcn_wmma_f32_16x16x32_f16(false, Ah, false, Bl, (short)0, c, false, false);
  c = __builtin_amdgcn_wmma_f32_16x16x32_f16(false, Al, false, Bh, (short)0, c, false, false);
  return c;
}

// ---------------------------------------------------------------------------
// Elementwise fp32 -> (hi fp16, lo fp16) split. 8 elements per thread.
// ---------------------------------------------------------------------------
__global__ __launch_bounds__(256) void conv_hilo_kernel(
    const float* __restrict__ X, _Float16* __restrict__ Xh,
    _Float16* __restrict__ Xl, int n8) {
  const int i = blockIdx.x * 256 + threadIdx.x;
  if (i >= n8) return;
  const float4 f0 = ((const float4*)X)[2 * i];
  const float4 f1 = ((const float4*)X)[2 * i + 1];
  const float xs[8] = {f0.x, f0.y, f0.z, f0.w, f1.x, f1.y, f1.z, f1.w};
  v8h h, l;
#pragma unroll
  for (int j = 0; j < 8; ++j) {
    _Float16 hh = (_Float16)xs[j];
    h[j] = hh;
    l[j] = (_Float16)(xs[j] - (float)hh);
  }
  ((v8h*)Xh)[i] = h;
  ((v8h*)Xl)[i] = l;
}

// ---------------------------------------------------------------------------
// GEMM: Y[M x 256] = X @ W^T + bias (X, W given as hi/lo fp16 pairs).
// One wave per 16x16 tile, K in 8 chunks of 32; 8 b128 loads + 3 WMMA/chunk.
// Epilogue optionally writes fp32 and/or hi/lo fp16 outputs.
// ---------------------------------------------------------------------------
__global__ __launch_bounds__(256) void gemm_bias_kernel(
    const _Float16* __restrict__ Xh, const _Float16* __restrict__ Xl,
    const _Float16* __restrict__ Wh, const _Float16* __restrict__ Wl,
    const float* __restrict__ bias,
    float* __restrict__ Yf, _Float16* __restrict__ Yh, _Float16* __restrict__ Yl,
    int act) {
  const int tid  = threadIdx.x;
  const int lane = tid & 31;
  const int gw   = blockIdx.x * 8 + (tid >> 5);   // global wave = tile index
  const int m0   = (gw >> 4) << 4;                // 16 N-tiles across DIM
  const int n0   = (gw & 15) << 4;
  const int hf   = lane >> 4;
  const int l    = lane & 15;

  const _Float16* xh = Xh + (size_t)(m0 + l) * DIM;
  const _Float16* xl = Xl + (size_t)(m0 + l) * DIM;
  const _Float16* wh = Wh + (size_t)(n0 + l) * DIM;
  const _Float16* wl = Wl + (size_t)(n0 + l) * DIM;

  v8f acc = {};
#pragma unroll
  for (int c = 0; c < 8; ++c) {
    const int ka = c * 32 + hf * 16;       // A: K = {ka..ka+15}
    const int kb = c * 32 + hf * 16;       // B: K = {kb..kb+15}
    v16h Ah = CAT16(*(const v8h*)(xh + ka), *(const v8h*)(xh + ka + 8));
    v16h Al = CAT16(*(const v8h*)(xl + ka), *(const v8h*)(xl + ka + 8));
    v16h Bh = CAT16(*(const v8h*)(wh + kb), *(const v8h*)(wh + kb + 8));
    v16h Bl = CAT16(*(const v8h*)(wl + kb), *(const v8h*)(wl + kb + 8));
    acc = wmma_hilo(Ah, Al, Bh, Bl, acc);
  }

  // C layout: row = m0 + r + hf*8, col = n0 + l
  const float bv = bias[n0 + l];
#pragma unroll
  for (int r = 0; r < 8; ++r) {
    float y = acc[r] + bv;
    if (act) y = 1.0f / (1.0f + __expf(-y));
    const size_t idx = (size_t)(m0 + hf * 8 + r) * DIM + n0 + l;
    if (Yf) Yf[idx] = y;
    if (Yh) {
      _Float16 h = (_Float16)y;
      Yh[idx] = h;
      Yl[idx] = (_Float16)(y - (float)h);
    }
  }
}

// ---------------------------------------------------------------------------
// Windowed attention. One 128-thread block per (batch, 64-query tile).
// Q/K arrive as hi/lo fp16; V is fp32. K window + V window staged in LDS
// (OOB rows zero-filled -> exactly reproduces reference's score*validf
// semantics: exp(0)=1 in denominator, zeroed vw). Each wave: 16 queries,
// 16x80 score band via WMMA, cross-lane softmax, packed-fp32 P*V.
// Output written directly as hi/lo fp16 (consumed only by the next GEMM).
// ---------------------------------------------------------------------------
__global__ __launch_bounds__(128) void attn_kernel(
    const _Float16* __restrict__ Qh, const _Float16* __restrict__ Ql,
    const _Float16* __restrict__ Kh, const _Float16* __restrict__ Kl,
    const float* V, _Float16* __restrict__ Oh, _Float16* __restrict__ Ol) {
  extern __shared__ char smem[];
  _Float16* Khs = (_Float16*)smem;                       // KROWS*KSTH halves
  _Float16* Kls = Khs + KROWS * KSTH;                    // KROWS*KSTH halves
  float*    Vs  = (float*)(smem + (size_t)2 * KROWS * KSTH * sizeof(_Float16));
  float*    Ps  = Vs + KROWS * VST;                      // 4 waves * 16 * 80

  const int tid = threadIdx.x;
  const int b   = blockIdx.x >> 6;
  const int q0  = (blockIdx.x & 63) << 6;
  const size_t base = (size_t)b * SEQ * DIM;
  const int r0 = q0 - WIN;

  // ---- Stage K hi/lo (fp16) rows [q0-32, q0+95], zero-fill OOB ----
  for (int i = tid; i < KROWS * (DIM / 8); i += 128) {
    const int t = i >> 5;                 // window row
    const int c = (i & 31) << 3;          // half column, step 8
    const int g = r0 + t;
    v8h kh = {}, kl = {};
    if (g >= 0 && g < SEQ) {
      kh = *(const v8h*)(Kh + base + (size_t)g * DIM + c);
      kl = *(const v8h*)(Kl + base + (size_t)g * DIM + c);
    }
    *(v8h*)(Khs + t * KSTH + c) = kh;
    *(v8h*)(Kls + t * KSTH + c) = kl;
  }
  // ---- Stage V (fp32) ----
  for (int i = tid; i < KROWS * (DIM / 4); i += 128) {
    const int t = i >> 6;
    const int c = (i & 63) << 2;
    const int g = r0 + t;
    float4 vv = {0.f, 0.f, 0.f, 0.f};
    if (g >= 0 && g < SEQ)
      vv = *(const float4*)(V + base + (size_t)g * DIM + c);
    *(float4*)(Vs + t * VST + c) = vv;
  }
  __syncthreads();

  const int lane = tid & 31;
  const int w    = tid >> 5;              // wave -> queries q0+16w..+15
  const int hf   = lane >> 4;
  const int l    = lane & 15;
  const int tw   = w * 16;
  const _Float16* qh = Qh + base + (size_t)(q0 + tw + l) * DIM;
  const _Float16* ql = Ql + base + (size_t)(q0 + tw + l) * DIM;

  // ---- Scores: 5 WMMA tiles x 8 K-chunks, pure loads + WMMA ----
  v8f acc[5] = {};
  for (int c = 0; c < 8; ++c) {
    const int ka = c * 32 + hf * 16;
    const int kb = c * 32 + hf * 16;
    v16h Ah = CAT16(*(const v8h*)(qh + ka), *(const v8h*)(qh + ka + 8));
    v16h Al = CAT16(*(const v8h*)(ql + ka), *(const v8h*)(ql + ka + 8));
#pragma unroll
    for (int tn = 0; tn < 5; ++tn) {
      const _Float16* kp = Khs + (size_t)(tw + tn * 16 + l) * KSTH + kb;
      const _Float16* kq = Kls + (size_t)(tw + tn * 16 + l) * KSTH + kb;
      v16h Bh = CAT16(*(const v8h*)kp, *(const v8h*)(kp + 8));
      v16h Bl = CAT16(*(const v8h*)kq, *(const v8h*)(kq + 8));
      acc[tn] = wmma_hilo(Ah, Al, Bh, Bl, acc[tn]);
    }
  }

  // ---- Softmax over the 65 in-window keys (band j in [m, m+64]) ----
  // C layout: acc[tn][r] at lane -> (row m = r + 8*hf, key j = 16*tn + l).
  float* Pw = Ps + w * 16 * 80;
#pragma unroll
  for (int r = 0; r < 8; ++r) {
    const int m = r + hf * 8;
    float mx = -1e30f;
#pragma unroll
    for (int tn = 0; tn < 5; ++tn) {
      const int j = tn * 16 + l;
      const bool in = (j >= m) && (j <= m + 64);
      float s = in ? acc[tn][r] * 0.0625f : -1e30f;   // 1/sqrt(256)
      acc[tn][r] = s;
      mx = fmaxf(mx, s);
    }
    mx = fmaxf(mx, __shfl_xor(mx, 1, 32));
    mx = fmaxf(mx, __shfl_xor(mx, 2, 32));
    mx = fmaxf(mx, __shfl_xor(mx, 4, 32));
    mx = fmaxf(mx, __shfl_xor(mx, 8, 32));
    float sum = 0.f;
#pragma unroll
    for (int tn = 0; tn < 5; ++tn) {
      const int j = tn * 16 + l;
      const bool in = (j >= m) && (j <= m + 64);
      float e = in ? __expf(acc[tn][r] - mx) : 0.f;
      acc[tn][r] = e;
      sum += e;
    }
    sum += __shfl_xor(sum, 1, 32);
    sum += __shfl_xor(sum, 2, 32);
    sum += __shfl_xor(sum, 4, 32);
    sum += __shfl_xor(sum, 8, 32);
    const float inv = 1.0f / sum;
#pragma unroll
    for (int tn = 0; tn < 5; ++tn)
      Pw[m * 80 + tn * 16 + l] = acc[tn][r] * inv;
  }
  __syncthreads();

  // ---- O[m][d] = sum_j P[m][j] * Vs[tw+j][d] ----
  // j-outer: V rows loaded once per key (2 b128), P broadcast as b128 per
  // 4 keys, all 16 query rows accumulated in v2f regs -> v_pk_fma_f32.
  const int d0 = lane << 2;               // dims 4*lane and 4*lane+128
  const v2f zz = {0.f, 0.f};
  v2f acc2[16][4];
#pragma unroll
  for (int m = 0; m < 16; ++m)
#pragma unroll
    for (int q = 0; q < 4; ++q) acc2[m][q] = zz;

  for (int j4 = 0; j4 < 80; j4 += 4) {
    float4 pv[16];
#pragma unroll
    for (int m = 0; m < 16; ++m)
      pv[m] = *(const float4*)(Pw + m * 80 + j4);   // broadcast read
#pragma unroll
    for (int jj = 0; jj < 4; ++jj) {
      const float* vr = Vs + (size_t)(tw + j4 + jj) * VST;
      const float4 v0 = *(const float4*)(vr + d0);
      const float4 v1 = *(const float4*)(vr + 128 + d0);
      const v2f va = {v0.x, v0.y}, vb = {v0.z, v0.w};
      const v2f vc = {v1.x, v1.y}, vd = {v1.z, v1.w};
#pragma unroll
      for (int m = 0; m < 16; ++m) {
        const float p = (jj == 0) ? pv[m].x : (jj == 1) ? pv[m].y
                      : (jj == 2) ? pv[m].z : pv[m].w;
        const v2f pp = {p, p};
        acc2[m][0] += pp * va;
        acc2[m][1] += pp * vb;
        acc2[m][2] += pp * vc;
        acc2[m][3] += pp * vd;
      }
    }
  }

  // ---- Emit hi/lo fp16 ----
#pragma unroll
  for (int m = 0; m < 16; ++m) {
    const float o[8] = {acc2[m][0][0], acc2[m][0][1], acc2[m][1][0], acc2[m][1][1],
                        acc2[m][2][0], acc2[m][2][1], acc2[m][3][0], acc2[m][3][1]};
    v4h h0, l0, h1, l1;
#pragma unroll
    for (int j = 0; j < 4; ++j) {
      _Float16 h = (_Float16)o[j];
      h0[j] = h; l0[j] = (_Float16)(o[j] - (float)h);
      _Float16 g = (_Float16)o[4 + j];
      h1[j] = g; l1[j] = (_Float16)(o[4 + j] - (float)g);
    }
    const size_t ro = base + (size_t)(q0 + tw + m) * DIM;
    *(v4h*)(Oh + ro + d0)       = h0;
    *(v4h*)(Ol + ro + d0)       = l0;
    *(v4h*)(Oh + ro + 128 + d0) = h1;
    *(v4h*)(Ol + ro + 128 + d0) = l1;
  }
}

// ---------------------------------------------------------------------------
extern "C" void kernel_launch(void* const* d_in, const int* in_sizes, int n_in,
                              void* d_out, int out_size, void* d_ws, size_t ws_size,
                              hipStream_t stream) {
  (void)in_sizes; (void)n_in; (void)out_size; (void)ws_size;
  const float* Q  = (const float*)d_in[0];
  const float* K  = (const float*)d_in[1];
  const float* V  = (const float*)d_in[2];
  const float* W  = (const float*)d_in[3];
  const float* bi = (const float*)d_in[4];
  float* out = (float*)d_out;

  const size_t N = (size_t)BATCH * SEQ * DIM;   // 4,194,304 elements
  // Workspace layout: 4 fp16 buffers (8 MB), 1 fp32 buffer (16 MB), W h/l.
  char* ws = (char*)d_ws;
  _Float16* h0 = (_Float16*)ws;                 // 8 MB each
  _Float16* h1 = h0 + N;
  _Float16* h2 = h1 + N;
  _Float16* h3 = h2 + N;
  float*    fb = (float*)(h3 + N);              // 16 MB fp32 (also fh0/fh1)
  _Float16* fh0 = (_Float16*)fb;
  _Float16* fh1 = fh0 + N;
  _Float16* wh = (_Float16*)((char*)fb + N * sizeof(float));
  _Float16* wl = wh + DIM * DIM;

  const dim3 agrd(BATCH * (SEQ / 64));          // 256 blocks
  const dim3 ablk(128);
  const size_t shmem = (size_t)2 * KROWS * KSTH * sizeof(_Float16)
                     + (size_t)KROWS * VST * sizeof(float)
                     + (size_t)4 * 16 * 80 * sizeof(float);   // 288,768 B
  const dim3 ggrd(((BATCH * SEQ) / 16) * (DIM / 16) / 8);     // 2048 blocks
  const dim3 gblk(256);

  // One-time hi/lo decomposition of external fp32 inputs.
  conv_hilo_kernel<<<(DIM * DIM / 8 + 255) / 256, 256, 0, stream>>>(W, wh, wl, DIM * DIM / 8);
  conv_hilo_kernel<<<(int)(N / 8 / 256), 256, 0, stream>>>(Q, h0, h1, (int)(N / 8));
  conv_hilo_kernel<<<(int)(N / 8 / 256), 256, 0, stream>>>(K, h2, h3, (int)(N / 8));

  // a1 = attn(query, key, value)          -> fh0/fh1 (hi/lo fp16)
  attn_kernel<<<agrd, ablk, shmem, stream>>>(h0, h1, h2, h3, V, fh0, fh1);
  // out1 = sigmoid(a1 @ W^T + b)          -> h0/h1 (hi/lo fp16)
  gemm_bias_kernel<<<ggrd, gblk, 0, stream>>>(fh0, fh1, wh, wl, bi, nullptr, h0, h1, 1);
  // proj = out1 @ W^T + b                 -> fb (fp32) + h2/h3 (hi/lo fp16)
  gemm_bias_kernel<<<ggrd, gblk, 0, stream>>>(h0, h1, wh, wl, bi, fb, h2, h3, 0);
  // a2 = attn(proj, proj, proj)           -> h0/h1 (hi/lo fp16)
  attn_kernel<<<agrd, ablk, shmem, stream>>>(h2, h3, h2, h3, fb, h0, h1);
  // out = sigmoid(a2 @ W^T + b)           -> d_out (fp32)
  gemm_bias_kernel<<<ggrd, gblk, 0, stream>>>(h0, h1, wh, wl, bi, out, nullptr, nullptr, 1);
}
